// Social_Aggregator_14422500180543
// MI455X (gfx1250) — matrix operand order repaired and checked by reading, not compiled, not run on hardware
//
#include <hip/hip_runtime.h>
#include <hip/hip_bf16.h>

typedef _Float16 v16h __attribute__((ext_vector_type(16)));
typedef _Float16 v8h  __attribute__((ext_vector_type(8)));
typedef float    v8f  __attribute__((ext_vector_type(8)));

#define KNB 50
#define DD  64
#define RR  8

// Fragment bases (units of 512 halves = one 32x16 B-fragment per 32 lanes x 16 halves)
#define W1_FRAG 0    // Kpad=96  -> 3 kc * 4 nt = 12 frags
#define W2_FRAG 12   // Kpad=64  -> 8 frags
#define A1_FRAG 20   // Kpad=128 -> 16 frags
#define A2_FRAG 36   // Kpad=64  -> 8 frags
#define NFRAGS  44

// ---------------------------------------------------------------------------
// Pre-swizzle the four weight matrices into f16 B-fragments.
// B-fragment layout used here (contiguous k, same order as A in wave_gemm): lane holds column c = nt*16 + (lane&15),
// halves i=0..15 hold K = kc*32 + (lane>>4)*16 + i.
// ---------------------------------------------------------------------------
__global__ __launch_bounds__(512) void pack_weights_kernel(
    const float* __restrict__ w1, const float* __restrict__ w2,
    const float* __restrict__ a1, const float* __restrict__ a2,
    _Float16* __restrict__ out)
{
  int f    = blockIdx.x;
  int tid  = threadIdx.x;
  int lane = tid >> 4;      // 0..31
  int i    = tid & 15;      // half index within lane
  const float* W; int Kdim; int lf;
  if (f < 12)      { W = w1; Kdim = 72;  lf = f;      }
  else if (f < 20) { W = w2; Kdim = 64;  lf = f - 12; }
  else if (f < 36) { W = a1; Kdim = 128; lf = f - 20; }
  else             { W = a2; Kdim = 64;  lf = f - 36; }
  int kc = lf >> 2, nt = lf & 3;
  int k  = kc*32 + (lane >> 4)*16 + i;
  int c  = nt*16 + (lane & 15);
  float v = (k < Kdim) ? W[k*64 + c] : 0.0f;
  out[f*512 + lane*16 + i] = (_Float16)v;
}

// ---------------------------------------------------------------------------
// One wave computes a 16x64 tile: 4 N-tiles, NKC K-chunks of 32.
// A layout used here (contiguous k, same order as packed B): lane r = lane&15, halves 0..15 = K kh*16+0..15
// where kh = lane>>4.
// ---------------------------------------------------------------------------
template<int KPAD, int NKC>
__device__ inline void wave_gemm(const _Float16* __restrict__ sA,
                                 const _Float16* __restrict__ wf,
                                 int mtile, int lane, v8f acc[4])
{
  const int r  = mtile*16 + (lane & 15);
  const int kh = (lane >> 4);
  #pragma unroll
  for (int kc = 0; kc < NKC; ++kc) {
    const _Float16* pa = sA + r*KPAD + kc*32 + kh*16;
    v8h lo = *(const v8h*)(pa);        // ds_load_b128
    v8h hi = *(const v8h*)(pa + 8);    // ds_load_b128
    v16h a = __builtin_shufflevector(lo, hi,
        0,1,2,3,4,5,6,7,8,9,10,11,12,13,14,15);
    #pragma unroll
    for (int nt = 0; nt < 4; ++nt) {
      v16h b = *(const v16h*)(wf + (kc*4 + nt)*512 + lane*16); // L2-resident
      acc[nt] = __builtin_amdgcn_wmma_f32_16x16x32_f16(
          false, a, false, b, (short)0, acc[nt], false, false);
    }
  }
}

// ---------------------------------------------------------------------------
// Main kernel: one workgroup (4 waves) per node b.
// ---------------------------------------------------------------------------
__global__ __launch_bounds__(128) void sagg_kernel(
    const int*   __restrict__ nodes,
    const int*   __restrict__ neigh_idx,
    const int*   __restrict__ neigh_len,
    const float* __restrict__ labels,
    const float* __restrict__ u2e,
    const float* __restrict__ w1_b, const float* __restrict__ w2_b,
    const float* __restrict__ a1_b, const float* __restrict__ a2_b,
    const float* __restrict__ a3_W, const float* __restrict__ a3_b,
    const _Float16* __restrict__ wf,
    float* __restrict__ out)
{
  __shared__ __attribute__((aligned(16))) _Float16 sA1[64*96];   // x  (K=72 pad 96)
  __shared__ __attribute__((aligned(16))) _Float16 sA2[64*64];   // h / a1-out
  __shared__ __attribute__((aligned(16))) _Float16 sA3[64*128];  // [o | u_rep]
  __shared__ float sO[64*64];
  __shared__ float sU[64];
  __shared__ float sScores[64];
  __shared__ float sAtt[64];
  __shared__ int   sIdx[64];
  __shared__ int   sLen;

  const int b     = blockIdx.x;
  const int tid   = threadIdx.x;
  const int lane  = tid & 31;
  const int mtile = tid >> 5;

  if (tid < KNB) sIdx[tid] = neigh_idx[b*KNB + tid];
  if (tid == 0)  sLen = neigh_len[b];
  if (tid < DD) {
    sU[tid]      = u2e[(long)nodes[b]*DD + tid];
    sScores[tid] = a3_b[0];
  }
  __syncthreads();

  // Stage x = [e_u | labels | 0-pad] into sA1 (f16), gathers hit L2.
  for (int e = tid; e < 64*96; e += 128) {
    int r = e / 96, c = e - r*96;
    float v = 0.0f;
    if (r < KNB) {
      if (c < DD)            v = u2e[(long)sIdx[r]*DD + c];
      else if (c < DD + RR)  v = labels[((long)b*KNB + r)*RR + (c - DD)];
    }
    sA1[e] = (_Float16)v;
  }
  // Broadcast u_rep into right half of att_in.
  for (int e = tid; e < 64*64; e += 128) {
    int r = e >> 6, c = e & 63;
    sA3[r*128 + 64 + c] = (_Float16)sU[c];
  }
  __syncthreads();

  // GEMM1: h = relu(x @ w1 + b1)   [64x96]x[96x64]
  {
    v8f acc[4] = {};
    wave_gemm<96,3>(sA1, wf + W1_FRAG*512, mtile, lane, acc);
    #pragma unroll
    for (int nt = 0; nt < 4; ++nt) {
      int col = nt*16 + (lane & 15);
      float bias = w1_b[col];
      #pragma unroll
      for (int v = 0; v < 8; ++v) {
        int row = mtile*16 + v + 8*(lane >> 4);
        float h = acc[nt][v] + bias;
        sA2[row*64 + col] = (_Float16)(h > 0.0f ? h : 0.0f);
      }
    }
  }
  __syncthreads();

  // GEMM2: o = relu(h @ w2 + b2); keep f32 copy for aggregation.
  {
    v8f acc[4] = {};
    wave_gemm<64,2>(sA2, wf + W2_FRAG*512, mtile, lane, acc);
    #pragma unroll
    for (int nt = 0; nt < 4; ++nt) {
      int col = nt*16 + (lane & 15);
      float bias = w2_b[col];
      #pragma unroll
      for (int v = 0; v < 8; ++v) {
        int row = mtile*16 + v + 8*(lane >> 4);
        float o = acc[nt][v] + bias;
        o = o > 0.0f ? o : 0.0f;
        sO[row*64 + col]   = o;
        sA3[row*128 + col] = (_Float16)o;
      }
    }
  }
  __syncthreads();

  // GEMM3: a = relu(att_in @ a1 + b)   [64x128]x[128x64]
  {
    v8f acc[4] = {};
    wave_gemm<128,4>(sA3, wf + A1_FRAG*512, mtile, lane, acc);
    #pragma unroll
    for (int nt = 0; nt < 4; ++nt) {
      int col = nt*16 + (lane & 15);
      float bias = a1_b[col];
      #pragma unroll
      for (int v = 0; v < 8; ++v) {
        int row = mtile*16 + v + 8*(lane >> 4);
        float h = acc[nt][v] + bias;
        sA2[row*64 + col] = (_Float16)(h > 0.0f ? h : 0.0f);
      }
    }
  }
  __syncthreads();

  // GEMM4 + scores: relu(a @ a2 + b) dotted with a3_W, reduced via LDS atomics.
  {
    v8f acc[4] = {};
    wave_gemm<64,2>(sA2, wf + A2_FRAG*512, mtile, lane, acc);
    float bias4[4], a3w4[4];
    #pragma unroll
    for (int nt = 0; nt < 4; ++nt) {
      int col = nt*16 + (lane & 15);
      bias4[nt] = a2_b[col];
      a3w4[nt]  = a3_W[col];
    }
    #pragma unroll
    for (int v = 0; v < 8; ++v) {
      float p = 0.0f;
      #pragma unroll
      for (int nt = 0; nt < 4; ++nt) {
        float val = acc[nt][v] + bias4[nt];
        val = val > 0.0f ? val : 0.0f;
        p += val * a3w4[nt];
      }
      atomicAdd(&sScores[mtile*16 + v + 8*(lane >> 4)], p);
    }
  }
  __syncthreads();

  // Masked softmax over neighbors (tiny: <=50 elements).
  if (tid == 0) {
    int len = sLen;
    float m = -3.0e38f;
    for (int k = 0; k < len; ++k) m = fmaxf(m, sScores[k]);
    float s = 0.0f;
    for (int k = 0; k < len; ++k) { float e = __expf(sScores[k] - m); sAtt[k] = e; s += e; }
    float inv = (s > 0.0f) ? 1.0f / s : 0.0f;
    for (int k = 0; k < len; ++k) sAtt[k] *= inv;
    for (int k = len; k < KNB; ++k) sAtt[k] = 0.0f;
  }
  __syncthreads();

  // Weighted aggregate; zero-neighbor fallback to own embedding.
  if (tid < DD) {
    float r;
    if (sLen > 0) {
      float sum = 0.0f;
      for (int k = 0; k < KNB; ++k) sum += sAtt[k] * sO[k*64 + tid];
      r = sum;
    } else {
      r = sU[tid];
    }
    out[(long)b*DD + tid] = r;
  }
}

extern "C" void kernel_launch(void* const* d_in, const int* in_sizes, int n_in,
                              void* d_out, int out_size, void* d_ws, size_t ws_size,
                              hipStream_t stream) {
  const int*   nodes     = (const int*)d_in[0];
  const int*   neigh_idx = (const int*)d_in[1];
  const int*   neigh_len = (const int*)d_in[2];
  const float* labels    = (const float*)d_in[3];
  const float* u2e       = (const float*)d_in[4];
  const float* w1_W      = (const float*)d_in[5];
  const float* w1_b      = (const float*)d_in[6];
  const float* w2_W      = (const float*)d_in[7];
  const float* w2_b      = (const float*)d_in[8];
  const float* a1_W      = (const float*)d_in[9];
  const float* a1_b      = (const float*)d_in[10];
  const float* a2_W      = (const float*)d_in[11];
  const float* a2_b      = (const float*)d_in[12];
  const float* a3_W      = (const float*)d_in[13];
  const float* a3_b      = (const float*)d_in[14];

  _Float16* wf = (_Float16*)d_ws;      // 44*512 halves = 45056 bytes of scratch
  const int B = in_sizes[0];

  pack_weights_kernel<<<NFRAGS, 512, 0, stream>>>(w1_W, w2_W, a1_W, a2_W, wf);
  sagg_kernel<<<B, 128, 0, stream>>>(nodes, neigh_idx, neigh_len, labels, u2e,
                                     w1_b, w2_b, a1_b, a2_b, a3_W, a3_b,
                                     wf, (float*)d_out);
}
